// TestAllReduceRMSNormGroupQuantFP8PackedModel_52647709114890
// MI455X (gfx1250) — matrix operand order repaired and checked by reading, not compiled, not run on hardware
//
#include <hip/hip_runtime.h>
#include <hip/hip_bf16.h>

typedef __attribute__((ext_vector_type(16))) __bf16 v16bf;
typedef __attribute__((ext_vector_type(8)))  __bf16 v8bf;
typedef __attribute__((ext_vector_type(8)))  float  v8f;

static constexpr int   T_TOK   = 16384;
static constexpr int   HID     = 2048;
static constexpr float EPS     = 1e-5f;
static constexpr float FP8_MAX = 448.0f;

// ---------------- gfx1250 async-to-LDS support (guarded) ----------------

#if defined(__gfx1250__) && __has_builtin(__builtin_amdgcn_global_load_async_to_lds_b128)
#define HAVE_ASYNC_LDS 1
typedef int v4i_gcc __attribute__((vector_size(16)));
typedef __attribute__((address_space(1))) v4i_gcc* as1_v4i_ptr;
typedef __attribute__((address_space(3))) v4i_gcc* as3_v4i_ptr;
__device__ __forceinline__ void async_load_b128(void* lds_dst, const void* gsrc) {
  __builtin_amdgcn_global_load_async_to_lds_b128(
      (as1_v4i_ptr)(v4i_gcc*)(void*)gsrc,
      (as3_v4i_ptr)(v4i_gcc*)lds_dst, 0, 0);
}
#else
#define HAVE_ASYNC_LDS 0
#endif

__device__ __forceinline__ void wait_async_zero() {
#if defined(__gfx1250__)
#if __has_builtin(__builtin_amdgcn_s_wait_asynccnt)
  __builtin_amdgcn_s_wait_asynccnt(0);
#else
  asm volatile("s_wait_asynccnt 0x0" ::: "memory");
#endif
#endif
}

// ---------------- helpers ----------------

__device__ __forceinline__ float warp_sum32(float v) {
#pragma unroll
  for (int off = 16; off > 0; off >>= 1) v += __shfl_xor(v, off, 32);
  return v;
}

__device__ __forceinline__ float block_sum256(float v, float* red) {
  v = warp_sum32(v);
  const int wave = threadIdx.x >> 5;
  if ((threadIdx.x & 31) == 0) red[wave] = v;
  __syncthreads();
  float t = 0.f;
#pragma unroll
  for (int i = 0; i < 8; ++i) t += red[i];
  return t;
}

// round-to-nearest-even to fp8 e4m3fn grid (returns the fp8 *value* as fp32)
__device__ __forceinline__ float fp8_e4m3_round(float v) {
  float av = fminf(fabsf(v), FP8_MAX);
  float r;
  if (av < 0.015625f) {                    // subnormal region: grid of 2^-9
    r = rintf(av * 512.0f) * (1.0f / 512.0f);
  } else {                                 // normal: RNE to 3 mantissa bits
    unsigned u   = __float_as_uint(av);
    unsigned lsb = (u >> 20) & 1u;
    u += 0x7FFFFu + lsb;
    u &= 0xFFF00000u;
    r = fminf(__uint_as_float(u), FP8_MAX);
  }
  return copysignf(r, v);
}

// ue8m0: smallest power of two >= max(amax,1e-10)/448
__device__ __forceinline__ float ue8m0_scale(float amax) {
  float s0 = fmaxf(amax, 1e-10f) * (1.0f / FP8_MAX);
  unsigned u    = __float_as_uint(s0);
  unsigned mant = u & 0x7FFFFFu;
  unsigned e    = (u >> 23) & 0xFFu;
  return (mant == 0u) ? s0 : __uint_as_float((e + 1u) << 23);
}

// ---------------- kernel 0: w fp32 [k][n] -> wT bf16 [n][k] ----------------

__global__ __launch_bounds__(256)
void convert_wT_kernel(const float* __restrict__ w, __bf16* __restrict__ wT) {
  __shared__ float tile[32][33];
  const int mat = blockIdx.z;
  const float* src = w  + (size_t)mat * HID * HID;
  __bf16*      dst = wT + (size_t)mat * HID * HID;
  const int n  = blockIdx.x * 32 + threadIdx.x;
  const int k0 = blockIdx.y * 32;
#pragma unroll
  for (int j = threadIdx.y; j < 32; j += 8)
    tile[j][threadIdx.x] = src[(size_t)(k0 + j) * HID + n];
  __syncthreads();
#pragma unroll
  for (int j = threadIdx.y; j < 32; j += 8)
    dst[(size_t)(blockIdx.x * 32 + j) * HID + k0 + threadIdx.x] =
        (__bf16)tile[threadIdx.x][j];
}

// ---------------- kernel 1: relu + rmsnorm -> bf16 act, fp32 resid ----------------

__global__ __launch_bounds__(256)
void stage_relu_norm_kernel(const float* __restrict__ x,
                            float* __restrict__ resid,
                            const float* __restrict__ nw,
                            __bf16* __restrict__ act) {
  __shared__ float red[8];
  const int token = blockIdx.x;
  const int base  = threadIdx.x * 8;
  const size_t idx = (size_t)token * HID + base;

  float v[8];
  const float4 a0 = ((const float4*)(x + idx))[0];
  const float4 a1 = ((const float4*)(x + idx))[1];
  v[0] = fmaxf(a0.x, 0.f); v[1] = fmaxf(a0.y, 0.f);
  v[2] = fmaxf(a0.z, 0.f); v[3] = fmaxf(a0.w, 0.f);
  v[4] = fmaxf(a1.x, 0.f); v[5] = fmaxf(a1.y, 0.f);
  v[6] = fmaxf(a1.z, 0.f); v[7] = fmaxf(a1.w, 0.f);

  float ss = 0.f;
#pragma unroll
  for (int i = 0; i < 8; ++i) ss += v[i] * v[i];
  const float total = block_sum256(ss, red);
  const float rs = rsqrtf(total * (1.0f / HID) + EPS);

  ((float4*)(resid + idx))[0] = make_float4(v[0], v[1], v[2], v[3]);
  ((float4*)(resid + idx))[1] = make_float4(v[4], v[5], v[6], v[7]);
#pragma unroll
  for (int i = 0; i < 8; ++i)
    act[idx + i] = (__bf16)(v[i] * rs * nw[base + i]);
}

// ---------------- kernel 2: bf16 WMMA GEMM  C[M,N] = A[M,K] * BT[N,K]^T ----------------
// block tile 128x256, 8 waves as 2(M)x4(N), each wave 64x64 = 4x4 WMMA tiles, K step 32.
// Staging: async global->LDS (double buffered) when available.

#define TILE_K     32
#define LDS_STRIDE 40          // 32 bf16 + 8 pad
#define ABUF_ELEMS (128 * LDS_STRIDE)
#define BBUF_ELEMS (256 * LDS_STRIDE)

__global__ __launch_bounds__(256)
void gemm_bf16_wmma_kernel(const __bf16* __restrict__ A,
                           const __bf16* __restrict__ BT,
                           float* __restrict__ C) {
  __shared__ __bf16 lA[2][ABUF_ELEMS];
  __shared__ __bf16 lB[2][BBUF_ELEMS];

  const int tid  = threadIdx.x;
  const int wave = tid >> 5;
  const int lane = tid & 31;
  const int row0 = blockIdx.y * 128;
  const int col0 = blockIdx.x * 256;

  // staging mapping: 256 threads; each stages 32B of A row (tid>>1) and 32B of
  // B rows (tid>>1) and (tid>>1)+128, half (tid&1).
  const int rr = tid >> 1;
  const int hh = tid & 1;
  const __bf16* gA  = A  + (size_t)(row0 + rr) * HID + hh * 16;
  const __bf16* gB0 = BT + (size_t)(col0 + rr) * HID + hh * 16;
  const __bf16* gB1 = BT + (size_t)(col0 + rr + 128) * HID + hh * 16;
  __bf16* dA  = &lA[0][rr * LDS_STRIDE + hh * 16];
  __bf16* dB0 = &lB[0][rr * LDS_STRIDE + hh * 16];
  __bf16* dB1 = &lB[0][(rr + 128) * LDS_STRIDE + hh * 16];

  // wave tile position
  const int wm    = (wave >> 2) * 64;   // 0 or 64
  const int wn    = (wave & 3) * 64;    // 0,64,128,192
  const int mrow  = lane & 15;
  const int kselA = (lane < 16) ? 0 : 8;    // A: lanes 0-15 hold K 0-7 & 16-23
  const int kselB = (lane < 16) ? 0 : 8;    // B: lanes 0-15 hold K 0-7 & 16-23

  v8f acc[4][4];
#pragma unroll
  for (int mt = 0; mt < 4; ++mt)
#pragma unroll
    for (int nt = 0; nt < 4; ++nt)
      acc[mt][nt] = (v8f){0.f, 0.f, 0.f, 0.f, 0.f, 0.f, 0.f, 0.f};

  auto compute_tile = [&](int buf) {
    const __bf16* bA = &lA[buf][0];
    const __bf16* bB = &lB[buf][0];
    union U16 { v16bf v; v8bf h[2]; };
    U16 ua[4];
#pragma unroll
    for (int mt = 0; mt < 4; ++mt) {
      const int m = wm + mt * 16 + mrow;
      ua[mt].h[0] = *(const v8bf*)&bA[m * LDS_STRIDE + kselA];       // K 0-7 / 8-15
      ua[mt].h[1] = *(const v8bf*)&bA[m * LDS_STRIDE + kselA + 16];  // K 16-23 / 24-31
    }
#pragma unroll
    for (int nt = 0; nt < 4; ++nt) {
      const int n = wn + nt * 16 + mrow;
      U16 ub;
      ub.h[0] = *(const v8bf*)&bB[n * LDS_STRIDE + kselB];
      ub.h[1] = *(const v8bf*)&bB[n * LDS_STRIDE + kselB + 16];
#pragma unroll
      for (int mt = 0; mt < 4; ++mt)
        acc[mt][nt] = __builtin_amdgcn_wmma_f32_16x16x32_bf16(
            false, ua[mt].v, false, ub.v, (short)0, acc[mt][nt], false, false);
    }
  };

#if HAVE_ASYNC_LDS
  auto issue_tile = [&](int buf, int kb) {
    const int ao = buf * ABUF_ELEMS;
    const int bo = buf * BBUF_ELEMS;
    async_load_b128(dA  + ao,     gA  + kb);
    async_load_b128(dA  + ao + 8, gA  + kb + 8);
    async_load_b128(dB0 + bo,     gB0 + kb);
    async_load_b128(dB0 + bo + 8, gB0 + kb + 8);
    async_load_b128(dB1 + bo,     gB1 + kb);
    async_load_b128(dB1 + bo + 8, gB1 + kb + 8);
  };

  issue_tile(0, 0);
  int buf = 0;
  for (int kb = 0; kb < HID; kb += TILE_K) {
    wait_async_zero();   // our async loads into `buf` have landed in LDS
    __syncthreads();     // everyone's landed; prev compute on buf^1 finished
    if (kb + TILE_K < HID) issue_tile(buf ^ 1, kb + TILE_K);
    compute_tile(buf);
    buf ^= 1;
  }
#else
  // synchronous fallback: register-staged, single LDS buffer
  uint4 ra0 = ((const uint4*)gA)[0],  ra1 = ((const uint4*)gA)[1];
  uint4 rb0 = ((const uint4*)gB0)[0], rb1 = ((const uint4*)gB0)[1];
  uint4 rc0 = ((const uint4*)gB1)[0], rc1 = ((const uint4*)gB1)[1];
  for (int kb = 0; kb < HID; kb += TILE_K) {
    __syncthreads();
    ((uint4*)dA)[0]       = ra0; *(uint4*)(dA + 8)  = ra1;
    ((uint4*)dB0)[0]      = rb0; *(uint4*)(dB0 + 8) = rb1;
    ((uint4*)dB1)[0]      = rc0; *(uint4*)(dB1 + 8) = rc1;
    __syncthreads();
    if (kb + TILE_K < HID) {
      const int kn = kb + TILE_K;
      ra0 = *(const uint4*)(gA + kn);  ra1 = *(const uint4*)(gA + kn + 8);
      rb0 = *(const uint4*)(gB0 + kn); rb1 = *(const uint4*)(gB0 + kn + 8);
      rc0 = *(const uint4*)(gB1 + kn); rc1 = *(const uint4*)(gB1 + kn + 8);
      __builtin_prefetch(gA + kn + TILE_K, 0, 1);
      __builtin_prefetch(gB0 + kn + TILE_K, 0, 1);
    }
    compute_tile(0);
  }
#endif

  // epilogue: C/D layout — VGPR i: lanes 0-15 M=i, lanes 16-31 M=i+8; N=lane&15
#pragma unroll
  for (int mt = 0; mt < 4; ++mt) {
#pragma unroll
    for (int nt = 0; nt < 4; ++nt) {
      const int n = col0 + wn + nt * 16 + (lane & 15);
#pragma unroll
      for (int i = 0; i < 8; ++i) {
        const int m = row0 + wm + mt * 16 + ((lane < 16) ? i : (i + 8));
        C[(size_t)m * HID + n] = acc[mt][nt][i];
      }
    }
  }
}

// ---------------- kernel 3: resid += x; y = rmsnorm(resid)*nw; optional ue8m0 quant ----------------

__global__ __launch_bounds__(256)
void stage_add_norm_kernel(const float* __restrict__ gin,
                           float* __restrict__ resid,
                           const float* __restrict__ nw,
                           __bf16* __restrict__ act,       // used if do_quant
                           float* __restrict__ scale_out,  // used if do_quant
                           float* __restrict__ y_out,      // used if !do_quant
                           int do_quant) {
  __shared__ float red[8];
  const int token = blockIdx.x;
  const int base  = threadIdx.x * 8;
  const size_t idx = (size_t)token * HID + base;

  float v[8];
  const float4 a0 = ((const float4*)(gin + idx))[0];
  const float4 a1 = ((const float4*)(gin + idx))[1];
  const float4 r0 = ((const float4*)(resid + idx))[0];
  const float4 r1 = ((const float4*)(resid + idx))[1];
  v[0] = a0.x + r0.x; v[1] = a0.y + r0.y; v[2] = a0.z + r0.z; v[3] = a0.w + r0.w;
  v[4] = a1.x + r1.x; v[5] = a1.y + r1.y; v[6] = a1.z + r1.z; v[7] = a1.w + r1.w;

  ((float4*)(resid + idx))[0] = make_float4(v[0], v[1], v[2], v[3]);
  ((float4*)(resid + idx))[1] = make_float4(v[4], v[5], v[6], v[7]);

  float ss = 0.f;
#pragma unroll
  for (int i = 0; i < 8; ++i) ss += v[i] * v[i];
  const float total = block_sum256(ss, red);
  const float rs = rsqrtf(total * (1.0f / HID) + EPS);

  float y[8];
#pragma unroll
  for (int i = 0; i < 8; ++i) y[i] = v[i] * rs * nw[base + i];

  if (do_quant) {
    float amax = 0.f;
#pragma unroll
    for (int i = 0; i < 8; ++i) amax = fmaxf(amax, fabsf(y[i]));
    // group = 128 elems = 16 consecutive lanes (groups never straddle a wave)
#pragma unroll
    for (int off = 8; off > 0; off >>= 1)
      amax = fmaxf(amax, __shfl_xor(amax, off, 16));
    const float scale = ue8m0_scale(amax);
    const float inv   = 1.0f / scale;          // exact: scale is a power of 2
    if ((threadIdx.x & 15) == 0)
      scale_out[(size_t)token * 16 + (threadIdx.x >> 4)] = scale;
#pragma unroll
    for (int i = 0; i < 8; ++i)
      act[idx + i] = (__bf16)fp8_e4m3_round(y[i] * inv);  // fp8 value exact in bf16
  } else {
    ((float4*)(y_out + idx))[0] = make_float4(y[0], y[1], y[2], y[3]);
    ((float4*)(y_out + idx))[1] = make_float4(y[4], y[5], y[6], y[7]);
  }
}

// ---------------- launch ----------------

extern "C" void kernel_launch(void* const* d_in, const int* in_sizes, int n_in,
                              void* d_out, int out_size, void* d_ws, size_t ws_size,
                              hipStream_t stream) {
  const float* x      = (const float*)d_in[0];   // [16384, 2048]
  const float* norm_w = (const float*)d_in[1];   // [4, 2048]
  const float* w      = (const float*)d_in[2];   // [3, 2048, 2048]

  float* out_y4 = (float*)d_out;                       // [16384, 2048]
  float* out_s2 = out_y4 + (size_t)T_TOK * HID;        // [16384, 16]
  float* out_s3 = out_s2 + (size_t)T_TOK * 16;         // [16384, 16]

  char* ws = (char*)d_ws;
  size_t off = 0;
  __bf16* wT    = (__bf16*)(ws + off); off += (size_t)3 * HID * HID * sizeof(__bf16);
  float*  resid = (float*) (ws + off); off += (size_t)T_TOK * HID * sizeof(float);
  __bf16* act   = (__bf16*)(ws + off); off += (size_t)T_TOK * HID * sizeof(__bf16);
  float*  gout  = (float*) (ws + off);

  const dim3 gemm_grid(HID / 256, T_TOK / 128);

  convert_wT_kernel<<<dim3(HID / 32, HID / 32, 3), dim3(32, 8), 0, stream>>>(w, wT);
  stage_relu_norm_kernel<<<T_TOK, 256, 0, stream>>>(x, resid, norm_w, act);

  gemm_bf16_wmma_kernel<<<gemm_grid, 256, 0, stream>>>(act, wT, gout);
  stage_add_norm_kernel<<<T_TOK, 256, 0, stream>>>(gout, resid, norm_w + HID,
                                                   act, out_s2, nullptr, 1);

  gemm_bf16_wmma_kernel<<<gemm_grid, 256, 0, stream>>>(act, wT + (size_t)HID * HID, gout);
  stage_add_norm_kernel<<<T_TOK, 256, 0, stream>>>(gout, resid, norm_w + 2 * HID,
                                                   act, out_s3, nullptr, 1);

  gemm_bf16_wmma_kernel<<<gemm_grid, 256, 0, stream>>>(act, wT + (size_t)2 * HID * HID, gout);
  stage_add_norm_kernel<<<T_TOK, 256, 0, stream>>>(gout, resid, norm_w + 3 * HID,
                                                   nullptr, nullptr, out_y4, 0);
}
